// LlamaDecoderLayer_88424786690713
// MI455X (gfx1250) — matrix operand order repaired and checked by reading, not compiled, not run on hardware
//
#include <hip/hip_runtime.h>
#include <hip/hip_bf16.h>

// ---------------------------------------------------------------------------
// Llama decoder layer for gfx1250 (MI455X): bf16 WMMA + TDM + async-LDS.
// T=2048 D=4096 N=32 K=8 H=128 F=14336
// ---------------------------------------------------------------------------

typedef __bf16 bf16_t;
typedef __attribute__((ext_vector_type(16))) __bf16    v16bf;
typedef __attribute__((ext_vector_type(8)))  float     v8f;
typedef __attribute__((ext_vector_type(4)))  unsigned  uint32x4;
typedef __attribute__((ext_vector_type(8)))  unsigned  uint32x8;

union FragBF {
    v16bf v;
    uint4 q[2];
};

__device__ __forceinline__ unsigned pack_bf2(float a, float b) {
    union { bf16_t h[2]; unsigned u; } x;
    x.h[0] = (bf16_t)a; x.h[1] = (bf16_t)b;
    return x.u;
}

// low 32 bits of a flat pointer to an LDS object == LDS byte address
__device__ __forceinline__ unsigned lds_addr_of(const void* p) {
    return (unsigned)(uintptr_t)p;
}

// Tensor Data Mover: DMA a 2D bf16 tile [tile_d1 rows x tile_d0 cols] from a
// row-major tensor (row stride = stride_d0 elements) into LDS, optionally
// padding pad_amt(+1) DWORDs after every (1<<pad_int) 8-byte units.
__device__ __forceinline__ void tdm_load_2d_bf16(
    unsigned lds_byte, const void* gptr,
    unsigned tensor_d0, unsigned tensor_d1,
    unsigned tile_d0, unsigned tile_d1,
    unsigned stride_d0,
    unsigned pad_int, unsigned pad_amt, int pad_en) {
    unsigned long long ga = (unsigned long long)(uintptr_t)gptr;
    uint32x4 g0;
    g0[0] = 1u;                                          // count=1, user mode
    g0[1] = lds_byte;                                    // lds_addr
    g0[2] = (unsigned)ga;                                // global_addr[31:0]
    g0[3] = ((unsigned)(ga >> 32) & 0x1FFFFFFu) | (2u << 30); // [56:32] | type=2
    uint32x8 g1;
    g1[0] = (1u << 16)                                   // data_size = 2 bytes
          | (pad_en ? ((1u << 20) | (pad_int << 22) | (pad_amt << 25)) : 0u);
    g1[1] = (tensor_d0 & 0xFFFFu) << 16;                 // tensor_dim0 lo
    g1[2] = (tensor_d0 >> 16) | ((tensor_d1 & 0xFFFFu) << 16);
    g1[3] = (tensor_d1 >> 16) | (tile_d0 << 16);         // tile_dim0
    g1[4] = tile_d1 & 0xFFFFu;                           // tile_dim1, tile_dim2=0
    g1[5] = stride_d0;                                   // tensor_dim0_stride lo
    g1[6] = 0u;
    g1[7] = 0u;
    asm volatile("tensor_load_to_lds %0, %1" :: "s"(g0), "s"(g1) : "memory");
}

__device__ __forceinline__ void async_load_b128_to_lds(unsigned lds_byte,
                                                       const void* gptr) {
    asm volatile("global_load_async_to_lds_b128 %0, %1, off"
                 :: "v"(lds_byte), "v"(gptr) : "memory");
}
__device__ __forceinline__ void wait_asynccnt0() {
    asm volatile("s_wait_asynccnt 0x0" ::: "memory");
}

constexpr int Tt = 2048;
constexpr int Dd = 4096;
constexpr int NH = 32;
constexpr int KH = 8;
constexpr int HD = 128;
constexpr int FF = 14336;

// ---------------------------------------------------------------------------
// RMSNorm + cast to bf16.  One block (256 threads) per row of D=4096.
// ---------------------------------------------------------------------------
__global__ __launch_bounds__(256)
void rmsnorm_cast_kernel(const float* __restrict__ x,
                         const float* __restrict__ scale,
                         bf16_t* __restrict__ out, int D) {
    int row = blockIdx.x;
    int t = threadIdx.x;
    const float* xr = x + (size_t)row * D;
    __shared__ float red[8];

    float4 v[4];
    float ss = 0.f;
#pragma unroll
    for (int i = 0; i < 4; ++i) {
        v[i] = ((const float4*)xr)[t * 4 + i];
        ss += v[i].x * v[i].x + v[i].y * v[i].y + v[i].z * v[i].z + v[i].w * v[i].w;
    }
#pragma unroll
    for (int m = 16; m >= 1; m >>= 1) ss += __shfl_xor(ss, m, 32);
    if ((t & 31) == 0) red[t >> 5] = ss;
    __syncthreads();
    float tot = 0.f;
#pragma unroll
    for (int i = 0; i < 8; ++i) tot += red[i];
    float r = rsqrtf(tot / (float)D + 1e-5f);

    unsigned o[8];
#pragma unroll
    for (int i = 0; i < 4; ++i) {
        float4 sv = ((const float4*)scale)[t * 4 + i];
        o[2 * i + 0] = pack_bf2(v[i].x * r * sv.x, v[i].y * r * sv.y);
        o[2 * i + 1] = pack_bf2(v[i].z * r * sv.z, v[i].w * r * sv.w);
    }
    uint4* op = (uint4*)(out + (size_t)row * D) + t * 2;
    op[0] = make_uint4(o[0], o[1], o[2], o[3]);
    op[1] = make_uint4(o[4], o[5], o[6], o[7]);
}

// ---------------------------------------------------------------------------
// RoPE applied in place to bf16 q [T,NH,HD] and k [T,KH,HD].
// ---------------------------------------------------------------------------
__global__ __launch_bounds__(64)
void rope_kernel(bf16_t* __restrict__ q, bf16_t* __restrict__ k,
                 const int* __restrict__ positions) {
    int t = blockIdx.x;
    int hd = blockIdx.y;
    int i = threadIdx.x;
    bf16_t* base;
    if (hd < NH) base = q + ((size_t)t * NH + hd) * HD;
    else         base = k + ((size_t)t * KH + (hd - NH)) * HD;
    float pos = (float)positions[t];
    float inv = __expf(-(float)i * (13.122363377404328f / 64.f)); // theta^(-i/64)
    float ang = pos * inv;
    float c, s;
    __sincosf(ang, &c, &s);
    float x1 = (float)base[i];
    float x2 = (float)base[i + 64];
    base[i]      = (bf16_t)(x1 * c - x2 * s);
    base[i + 64] = (bf16_t)(x2 * c + x1 * s);
}

// ---------------------------------------------------------------------------
// Tiled bf16 WMMA GEMM:  C[M,Nn] = A[M,Kk](bf16) * B[Kk,Nn](f32->bf16)
// 256 threads = 8 waves; 128x128 tile; wave tile 32x64 (2x4 WMMA tiles).
// A tile staged by the Tensor Data Mover (with LDS row padding 32->40 elems).
// ---------------------------------------------------------------------------
#define LDSA_S 40
#define LDSB_S 40

__global__ __launch_bounds__(256)
void gemm_bf16_kernel(const bf16_t* __restrict__ A, const float* __restrict__ B,
                      void* __restrict__ Cout, const float* __restrict__ Cadd,
                      int M, int Nn, int Kk, int store_f32) {
    __shared__ bf16_t ldsA[128 * LDSA_S];
    __shared__ bf16_t ldsB[128 * LDSB_S];

    int t = threadIdx.x;
    int wave = t >> 5, lane = t & 31;
    int l15 = lane & 15, g = lane >> 4;
    int m0 = blockIdx.x * 128, n0 = blockIdx.y * 128;
    int wm = (wave & 3) * 32, wn = (wave >> 2) * 64;
    int wave_s = __builtin_amdgcn_readfirstlane(wave);   // scalar wave id
    unsigned ldsA_byte = lds_addr_of(ldsA);

    v8f acc[2][4];
    v8f vz = {0.f, 0.f, 0.f, 0.f, 0.f, 0.f, 0.f, 0.f};
#pragma unroll
    for (int a = 0; a < 2; ++a)
#pragma unroll
        for (int b = 0; b < 4; ++b) acc[a][b] = vz;

    int bn = lane * 4, bk = (t >> 5) * 4;   // B staging: 4n x 4k per thread

    for (int kt = 0; kt < Kk; kt += 32) {
        // ---- stage B (f32 global -> regs) ----
        const float* bbase = B + (size_t)(kt + bk) * Nn + n0 + bn;
        float4 b0 = *(const float4*)(bbase);
        float4 b1 = *(const float4*)(bbase + (size_t)Nn);
        float4 b2 = *(const float4*)(bbase + (size_t)Nn * 2);
        float4 b3 = *(const float4*)(bbase + (size_t)Nn * 3);
        if (kt + 32 < Kk)
            __builtin_prefetch((const void*)(bbase + (size_t)Nn * 32), 0, 3);

        __syncthreads();   // previous tile fully consumed by all waves

        // ---- stage A tile via Tensor Data Mover (wave 0 only, scalar branch)
        if (wave_s == 0)
            tdm_load_2d_bf16(ldsA_byte, A + (size_t)m0 * Kk + kt,
                             (unsigned)Kk, (unsigned)M,
                             32u, 128u, (unsigned)Kk,
                             /*pad_int 16 dwords*/3u, /*pad 4 dwords*/3u, 1);

        // ---- convert + store B transposed (n-major, packed k pairs) ----
        *(uint2*)&ldsB[(bn + 0) * LDSB_S + bk] = make_uint2(pack_bf2(b0.x, b1.x), pack_bf2(b2.x, b3.x));
        *(uint2*)&ldsB[(bn + 1) * LDSB_S + bk] = make_uint2(pack_bf2(b0.y, b1.y), pack_bf2(b2.y, b3.y));
        *(uint2*)&ldsB[(bn + 2) * LDSB_S + bk] = make_uint2(pack_bf2(b0.z, b1.z), pack_bf2(b2.z, b3.z));
        *(uint2*)&ldsB[(bn + 3) * LDSB_S + bk] = make_uint2(pack_bf2(b0.w, b1.w), pack_bf2(b2.w, b3.w));

        __builtin_amdgcn_s_wait_tensorcnt(0);   // no-op for waves 1..7
        __syncthreads();

        // ---- fragments + WMMA ----
        FragBF af[2], bf[4];
#pragma unroll
        for (int tm = 0; tm < 2; ++tm) {
            int row = wm + 16 * tm + l15;
            af[tm].q[0] = *(const uint4*)&ldsA[row * LDSA_S + 8 * g];
            af[tm].q[1] = *(const uint4*)&ldsA[row * LDSA_S + 16 + 8 * g];
        }
#pragma unroll
        for (int tn = 0; tn < 4; ++tn) {
            int col = wn + 16 * tn + l15;
            bf[tn].q[0] = *(const uint4*)&ldsB[col * LDSB_S + 8 * g];
            bf[tn].q[1] = *(const uint4*)&ldsB[col * LDSB_S + 16 + 8 * g];
        }
#pragma unroll
        for (int tm = 0; tm < 2; ++tm)
#pragma unroll
            for (int tn = 0; tn < 4; ++tn)
                acc[tm][tn] = __builtin_amdgcn_wmma_f32_16x16x32_bf16(
                    false, af[tm].v, false, bf[tn].v, (short)0, acc[tm][tn], false, false);
    }

    // ---- epilogue ----
#pragma unroll
    for (int tm = 0; tm < 2; ++tm)
#pragma unroll
        for (int tn = 0; tn < 4; ++tn)
#pragma unroll
            for (int r = 0; r < 8; ++r) {
                int row = m0 + wm + 16 * tm + r + 8 * g;
                int col = n0 + wn + 16 * tn + l15;
                size_t idx = (size_t)row * Nn + col;
                float val = acc[tm][tn][r];
                if (store_f32) {
                    if (Cadd) val += Cadd[idx];
                    ((float*)Cout)[idx] = val;
                } else {
                    ((bf16_t*)Cout)[idx] = (bf16_t)val;
                }
            }
}

// ---------------------------------------------------------------------------
// Fused gate/up GEMM + SiLU:  gu = silu(A*Wg) * (A*Wu), bf16 out [M,Nn]
// ---------------------------------------------------------------------------
__global__ __launch_bounds__(256)
void mlp_gateup_kernel(const bf16_t* __restrict__ A, const float* __restrict__ Bg,
                       const float* __restrict__ Bu, bf16_t* __restrict__ Out,
                       int M, int Nn, int Kk) {
    __shared__ bf16_t ldsA[128 * LDSA_S];
    __shared__ bf16_t ldsBg[128 * LDSB_S];
    __shared__ bf16_t ldsBu[128 * LDSB_S];

    int t = threadIdx.x;
    int wave = t >> 5, lane = t & 31;
    int l15 = lane & 15, g = lane >> 4;
    int m0 = blockIdx.x * 128, n0 = blockIdx.y * 128;
    int wm = (wave & 3) * 32, wn = (wave >> 2) * 64;
    int wave_s = __builtin_amdgcn_readfirstlane(wave);
    unsigned ldsA_byte = lds_addr_of(ldsA);

    v8f accg[2][4], accu[2][4];
    v8f vz = {0.f, 0.f, 0.f, 0.f, 0.f, 0.f, 0.f, 0.f};
#pragma unroll
    for (int a = 0; a < 2; ++a)
#pragma unroll
        for (int b = 0; b < 4; ++b) { accg[a][b] = vz; accu[a][b] = vz; }

    int bn = lane * 4, bk = (t >> 5) * 4;

    for (int kt = 0; kt < Kk; kt += 32) {
        const float* gb = Bg + (size_t)(kt + bk) * Nn + n0 + bn;
        const float* ub = Bu + (size_t)(kt + bk) * Nn + n0 + bn;
        float4 g0 = *(const float4*)(gb);
        float4 g1 = *(const float4*)(gb + (size_t)Nn);
        float4 g2 = *(const float4*)(gb + (size_t)Nn * 2);
        float4 g3 = *(const float4*)(gb + (size_t)Nn * 3);
        float4 u0 = *(const float4*)(ub);
        float4 u1 = *(const float4*)(ub + (size_t)Nn);
        float4 u2 = *(const float4*)(ub + (size_t)Nn * 2);
        float4 u3 = *(const float4*)(ub + (size_t)Nn * 3);

        __syncthreads();
        if (wave_s == 0)
            tdm_load_2d_bf16(ldsA_byte, A + (size_t)m0 * Kk + kt,
                             (unsigned)Kk, (unsigned)M,
                             32u, 128u, (unsigned)Kk, 3u, 3u, 1);

        *(uint2*)&ldsBg[(bn + 0) * LDSB_S + bk] = make_uint2(pack_bf2(g0.x, g1.x), pack_bf2(g2.x, g3.x));
        *(uint2*)&ldsBg[(bn + 1) * LDSB_S + bk] = make_uint2(pack_bf2(g0.y, g1.y), pack_bf2(g2.y, g3.y));
        *(uint2*)&ldsBg[(bn + 2) * LDSB_S + bk] = make_uint2(pack_bf2(g0.z, g1.z), pack_bf2(g2.z, g3.z));
        *(uint2*)&ldsBg[(bn + 3) * LDSB_S + bk] = make_uint2(pack_bf2(g0.w, g1.w), pack_bf2(g2.w, g3.w));
        *(uint2*)&ldsBu[(bn + 0) * LDSB_S + bk] = make_uint2(pack_bf2(u0.x, u1.x), pack_bf2(u2.x, u3.x));
        *(uint2*)&ldsBu[(bn + 1) * LDSB_S + bk] = make_uint2(pack_bf2(u0.y, u1.y), pack_bf2(u2.y, u3.y));
        *(uint2*)&ldsBu[(bn + 2) * LDSB_S + bk] = make_uint2(pack_bf2(u0.z, u1.z), pack_bf2(u2.z, u3.z));
        *(uint2*)&ldsBu[(bn + 3) * LDSB_S + bk] = make_uint2(pack_bf2(u0.w, u1.w), pack_bf2(u2.w, u3.w));

        __builtin_amdgcn_s_wait_tensorcnt(0);
        __syncthreads();

        FragBF af[2];
#pragma unroll
        for (int tm = 0; tm < 2; ++tm) {
            int row = wm + 16 * tm + l15;
            af[tm].q[0] = *(const uint4*)&ldsA[row * LDSA_S + 8 * g];
            af[tm].q[1] = *(const uint4*)&ldsA[row * LDSA_S + 16 + 8 * g];
        }
#pragma unroll
        for (int tn = 0; tn < 4; ++tn) {
            int col = wn + 16 * tn + l15;
            FragBF bfg, bfu;
            bfg.q[0] = *(const uint4*)&ldsBg[col * LDSB_S + 8 * g];
            bfg.q[1] = *(const uint4*)&ldsBg[col * LDSB_S + 16 + 8 * g];
            bfu.q[0] = *(const uint4*)&ldsBu[col * LDSB_S + 8 * g];
            bfu.q[1] = *(const uint4*)&ldsBu[col * LDSB_S + 16 + 8 * g];
#pragma unroll
            for (int tm = 0; tm < 2; ++tm) {
                accg[tm][tn] = __builtin_amdgcn_wmma_f32_16x16x32_bf16(
                    false, af[tm].v, false, bfg.v, (short)0, accg[tm][tn], false, false);
                accu[tm][tn] = __builtin_amdgcn_wmma_f32_16x16x32_bf16(
                    false, af[tm].v, false, bfu.v, (short)0, accu[tm][tn], false, false);
            }
        }
    }

#pragma unroll
    for (int tm = 0; tm < 2; ++tm)
#pragma unroll
        for (int tn = 0; tn < 4; ++tn)
#pragma unroll
            for (int r = 0; r < 8; ++r) {
                int row = m0 + wm + 16 * tm + r + 8 * g;
                int col = n0 + wn + 16 * tn + l15;
                float gv = accg[tm][tn][r];
                float uv = accu[tm][tn][r];
                float silu = gv / (1.f + __expf(-gv));
                Out[(size_t)row * Nn + col] = (bf16_t)(silu * uv);
            }
}

// ---------------------------------------------------------------------------
// Flash attention, GQA (4 q-heads per kv-head), causal.
// One wave per (16-query tile, head); 32 KV tokens per iteration.
// K tile staged via async global->LDS copy; V tile transposed through LDS.
// ---------------------------------------------------------------------------
#define LDSK_S 136   // 128 + 8 pad, keeps 16B alignment

__global__ __launch_bounds__(32)
void attn_kernel(const bf16_t* __restrict__ q, const bf16_t* __restrict__ k,
                 const bf16_t* __restrict__ v, bf16_t* __restrict__ o) {
    __shared__ bf16_t ldsP[16 * 40];
    __shared__ bf16_t ldsVt[128 * 40];
    __shared__ bf16_t ldsK[32 * LDSK_S];

    int lane = threadIdx.x;
    int l15 = lane & 15, g = lane >> 4;
    int qb = blockIdx.x, head = blockIdx.y;
    int kvh = head >> 2;
    int q0 = qb * 16;
    unsigned ldsK_byte = lds_addr_of(ldsK);

    // Q fragments: 4 chunks of H (A-matrix layout)
    FragBF qf[4];
    const bf16_t* qrow = q + ((size_t)(q0 + l15) * NH + head) * HD;
#pragma unroll
    for (int c = 0; c < 4; ++c) {
        qf[c].q[0] = *(const uint4*)(qrow + 32 * c + 8 * g);
        qf[c].q[1] = *(const uint4*)(qrow + 32 * c + 16 + 8 * g);
    }

    float mrun[8], lrun[8];
    v8f accv[8];
    v8f vz = {0.f, 0.f, 0.f, 0.f, 0.f, 0.f, 0.f, 0.f};
#pragma unroll
    for (int r = 0; r < 8; ++r) { mrun[r] = -3.0e38f; lrun[r] = 0.f; }
#pragma unroll
    for (int c = 0; c < 8; ++c) accv[c] = vz;

    const float sc = 0.08838834764831845f; // 1/sqrt(128)

    for (int s0 = 0; s0 < q0 + 16; s0 += 32) {
        // ---- async-copy K tile (32 tokens x 128 dims) into LDS ----
#pragma unroll
        for (int i = 0; i < 16; ++i) {
            int row = 2 * i + g;                    // lanes cover 2 rows/issue
            unsigned daddr = ldsK_byte + (unsigned)(row * LDSK_S + l15 * 8) * 2u;
            const bf16_t* saddr =
                k + ((size_t)(s0 + row) * KH + kvh) * HD + l15 * 8;
            async_load_b128_to_lds(daddr, saddr);
        }

        // ---- stage V tile transposed into LDS (overlaps with async K) ----
#pragma unroll 4
        for (int kk = 0; kk < 32; ++kk) {
            union { uint2 u; bf16_t h[4]; } w;
            w.u = *(const uint2*)(v + ((size_t)(s0 + kk) * KH + kvh) * HD + 4 * lane);
#pragma unroll
            for (int j = 0; j < 4; ++j)
                ldsVt[(4 * lane + j) * 40 + kk] = w.h[j];
        }

        wait_asynccnt0();

        // ---- S = Q K^T (two 16x16 column tiles) ----
        v8f S[2] = {vz, vz};
#pragma unroll
        for (int j = 0; j < 2; ++j) {
            int krow = 16 * j + l15;
#pragma unroll
            for (int c = 0; c < 4; ++c) {
                FragBF kf;
                kf.q[0] = *(const uint4*)&ldsK[krow * LDSK_S + 32 * c + 8 * g];
                kf.q[1] = *(const uint4*)&ldsK[krow * LDSK_S + 32 * c + 16 + 8 * g];
                S[j] = __builtin_amdgcn_wmma_f32_16x16x32_bf16(
                    false, qf[c].v, false, kf.v, (short)0, S[j], false, false);
            }
        }

        // ---- online softmax (row r in VGPR r, lanes = columns) ----
        float p0[8], p1[8];
#pragma unroll
        for (int r = 0; r < 8; ++r) {
            int rowt = q0 + r + 8 * g;
            float s0v = S[0][r] * sc;
            float s1v = S[1][r] * sc;
            if (s0 + l15 > rowt)      s0v = -3.0e38f;
            if (s0 + 16 + l15 > rowt) s1v = -3.0e38f;
            float mx = fmaxf(s0v, s1v);
#pragma unroll
            for (int d = 1; d < 16; d <<= 1) mx = fmaxf(mx, __shfl_xor(mx, d, 16));
            float mn = fmaxf(mrun[r], mx);
            float al = __expf(mrun[r] - mn);
            float e0 = __expf(s0v - mn);
            float e1 = __expf(s1v - mn);
            float sum = e0 + e1;
#pragma unroll
            for (int d = 1; d < 16; d <<= 1) sum += __shfl_xor(sum, d, 16);
            lrun[r] = lrun[r] * al + sum;
            mrun[r] = mn;
            p0[r] = e0; p1[r] = e1;
#pragma unroll
            for (int c = 0; c < 8; ++c) accv[c][r] *= al;
        }

        // ---- transpose P through LDS into A-fragment layout ----
#pragma unroll
        for (int r = 0; r < 8; ++r) {
            int prow = r + 8 * g;
            ldsP[prow * 40 + l15]      = (bf16_t)p0[r];
            ldsP[prow * 40 + 16 + l15] = (bf16_t)p1[r];
        }
        __syncthreads();
        FragBF pf;
        pf.q[0] = *(const uint4*)&ldsP[l15 * 40 + 8 * g];
        pf.q[1] = *(const uint4*)&ldsP[l15 * 40 + 16 + 8 * g];

        // ---- O += P V ----
#pragma unroll
        for (int cc = 0; cc < 8; ++cc) {
            FragBF vf;
            vf.q[0] = *(const uint4*)&ldsVt[(16 * cc + l15) * 40 + 8 * g];
            vf.q[1] = *(const uint4*)&ldsVt[(16 * cc + l15) * 40 + 16 + 8 * g];
            accv[cc] = __builtin_amdgcn_wmma_f32_16x16x32_bf16(
                false, pf.v, false, vf.v, (short)0, accv[cc], false, false);
        }
        __syncthreads();
    }

    // ---- normalize + store ----
#pragma unroll
    for (int r = 0; r < 8; ++r) {
        float inv = 1.f / lrun[r];
        int rowt = q0 + r + 8 * g;
#pragma unroll
        for (int cc = 0; cc < 8; ++cc)
            o[((size_t)rowt * NH + head) * HD + 16 * cc + l15] =
                (bf16_t)(accv[cc][r] * inv);
    }
}

// ---------------------------------------------------------------------------
// Launcher
// ---------------------------------------------------------------------------
extern "C" void kernel_launch(void* const* d_in, const int* in_sizes, int n_in,
                              void* d_out, int out_size, void* d_ws, size_t ws_size,
                              hipStream_t stream) {
    const float* x       = (const float*)d_in[0];
    const int*   posits  = (const int*)  d_in[1];
    const float* ln1     = (const float*)d_in[2];
    const float* w_q     = (const float*)d_in[3];
    const float* w_k     = (const float*)d_in[4];
    const float* w_v     = (const float*)d_in[5];
    const float* w_o     = (const float*)d_in[6];
    const float* ln2     = (const float*)d_in[7];
    const float* w_gate  = (const float*)d_in[8];
    const float* w_up    = (const float*)d_in[9];
    const float* w_down  = (const float*)d_in[10];

    char* ws = (char*)d_ws;
    size_t off = 0;
    auto alloc = [&](size_t bytes) {
        char* p = ws + off;
        off += (bytes + 255) & ~(size_t)255;
        return p;
    };
    bf16_t* h     = (bf16_t*)alloc((size_t)Tt * Dd * 2);
    bf16_t* qb    = (bf16_t*)alloc((size_t)Tt * NH * HD * 2);
    bf16_t* kb    = (bf16_t*)alloc((size_t)Tt * KH * HD * 2);
    bf16_t* vb    = (bf16_t*)alloc((size_t)Tt * KH * HD * 2);
    bf16_t* attnb = (bf16_t*)alloc((size_t)Tt * NH * HD * 2);
    float*  resid = (float*) alloc((size_t)Tt * Dd * 4);
    bf16_t* h2    = (bf16_t*)alloc((size_t)Tt * Dd * 2);
    bf16_t* gu    = (bf16_t*)alloc((size_t)Tt * FF * 2);

    rmsnorm_cast_kernel<<<Tt, 256, 0, stream>>>(x, ln1, h, Dd);

    gemm_bf16_kernel<<<dim3(Tt / 128, (NH * HD) / 128), 256, 0, stream>>>(
        h, w_q, qb, nullptr, Tt, NH * HD, Dd, 0);
    gemm_bf16_kernel<<<dim3(Tt / 128, (KH * HD) / 128), 256, 0, stream>>>(
        h, w_k, kb, nullptr, Tt, KH * HD, Dd, 0);
    gemm_bf16_kernel<<<dim3(Tt / 128, (KH * HD) / 128), 256, 0, stream>>>(
        h, w_v, vb, nullptr, Tt, KH * HD, Dd, 0);

    rope_kernel<<<dim3(Tt, NH + KH), 64, 0, stream>>>(qb, kb, posits);

    attn_kernel<<<dim3(Tt / 16, NH), 32, 0, stream>>>(qb, kb, vb, attnb);

    gemm_bf16_kernel<<<dim3(Tt / 128, Dd / 128), 256, 0, stream>>>(
        attnb, w_o, resid, x, Tt, Dd, NH * HD, 1);

    rmsnorm_cast_kernel<<<Tt, 256, 0, stream>>>(resid, ln2, h2, Dd);

    mlp_gateup_kernel<<<dim3(Tt / 128, FF / 128), 256, 0, stream>>>(
        h2, w_gate, w_up, gu, Tt, FF, Dd);
    gemm_bf16_kernel<<<dim3(Tt / 128, Dd / 128), 256, 0, stream>>>(
        gu, w_down, d_out, resid, Tt, Dd, FF, 1);
}
